// DiffsolClassifier_58566174048881
// MI455X (gfx1250) — compile-verified, matrix operand order checked by reading, not run on hardware
//
#include <hip/hip_runtime.h>
#include <hip/hip_bf16.h>
#include <math.h>

// CDNA5 / gfx1250: wave32, WMMA bf16 16x16x32, TDM async tensor load.
typedef __attribute__((ext_vector_type(16))) __bf16 v16bf;
typedef __attribute__((ext_vector_type(8)))  __bf16 v8bf;
typedef __attribute__((ext_vector_type(8)))  float  v8f;
typedef __attribute__((ext_vector_type(4)))  unsigned int v4u;
typedef __attribute__((ext_vector_type(8)))  int  v8i;
typedef __attribute__((ext_vector_type(4)))  int  v4i;

union V16 { v16bf v; v8bf h[2]; };

#define IMGS 4          // images per workgroup (B=65536 divisible)
#define NTHREADS 256    // 8 waves of 32
#define NWAVES (NTHREADS / 32)

// conv1 GEMM: per-image M padded 196 -> 208 (13 tiles), K 9 -> 16, N 8 -> 16
// conv2 GEMM: per-image M padded  49 ->  64 ( 4 tiles), K 72 -> 96, N = 16
//   conv2 K order: k = (ky*3+kx)*8 + c  (channel minor -> contiguous b128 moves)

// 1-D contiguous TDM load: ndw dwords from gsrc -> LDS byte offset lds_off.
// D# per CDNA5 ISA ch.8: group0 {count, lds_addr, global_addr, type=2},
// group1 {data_size=4B, tensor_dim0=tile_dim0=ndw, dim1=1, stride0=ndw}.
// This toolchain exposes the 6-arg builtin: (v4u, v8i, v4i, v4i, v8i, i32 cpol).
__device__ __forceinline__ void tdm_load_1d(const void* gsrc, unsigned int lds_off,
                                            unsigned int ndw) {
    unsigned long long ga = (unsigned long long)(uintptr_t)gsrc;
    v4u g0;
    g0[0] = 1u;                                               // count=1, user mode
    g0[1] = lds_off;                                          // lds_addr [63:32]
    g0[2] = (unsigned int)ga;                                 // global_addr[31:0]
    g0[3] = (unsigned int)((ga >> 32) & 0x01FFFFFFu) | (2u << 30); // addr[56:32]|type=2
    v8i g1;
    g1[0] = (int)(2u << 16);                                  // data_size=2 -> 4 bytes
    g1[1] = (int)(ndw << 16);                                 // tensor_dim0[15:0]
    g1[2] = (int)((ndw >> 16) | (1u << 16));                  // dim0 hi | tensor_dim1 lo=1
    g1[3] = (int)(ndw << 16);                                 // dim1 hi=0 | tile_dim0
    g1[4] = 1;                                                // tile_dim1=1, tile_dim2=0
    g1[5] = (int)ndw;                                         // tensor_dim0_stride lo
    g1[6] = 0;
    g1[7] = 0;
    v4i z4 = {0, 0, 0, 0};
    v8i z8 = {0, 0, 0, 0, 0, 0, 0, 0};
    __builtin_amdgcn_tensor_load_to_lds(g0, g1, z4, z4, z8, 0);
}

__global__ __launch_bounds__(NTHREADS)
void diffsol_fused_kernel(const float* __restrict__ x,
                          const float* __restrict__ W1, const float* __restrict__ b1,
                          const float* __restrict__ W2, const float* __restrict__ b2,
                          const float* __restrict__ Wfc, const float* __restrict__ bfc,
                          float* __restrict__ out)
{
    // ---- LDS (~74 KB; WGP has 320 KB) ----
    __shared__ __align__(16) union {
        struct {
            float  xs[IMGS][29][32];        // zero-padded input: pixel (iy,ix) at [iy+1][ix+1]
            union {
                __bf16 im1[IMGS * 208 * 16];// conv1 im2col (rows 196..207/img junk)
                float  xr[IMGS * 784];      // TDM landing zone for raw x tile
            } m;
            __bf16 z1[IMGS][8][208];        // conv1 out, channel-major, relu bf16
        } a;                                 // 41472 B
        __bf16 im2[IMGS * 64 * 96];         // conv2 im2col (rows 49..63/img junk)  49152 B
    } u;
    __shared__ __align__(16) __bf16 z1p[IMGS][16][16][8];  // channel-last, zero border
    __shared__ float w1s[72], b1s[8];
    __shared__ float w2s[1152], b2s[16];
    __shared__ float wfcs[784];
    __shared__ float bfcs;
    __shared__ float partial[IMGS];

    const int tid  = threadIdx.x;
    const int lane = tid & 31;
    const int wave = tid >> 5;
    const int n    = lane & 15;        // WMMA column this lane owns
    const int hiK  = (lane >> 4) * 8;  // lanes 16-31: +8 K chunk (A layout)
    const int hiM  = (lane >> 4) * 8;  // lanes 16-31: +8 M rows (C/D layout)

    // ---- Phase 0a: kick off TDM for x tile; zero padded buffers; stage weights ----
    if (wave == 0) {
        tdm_load_1d(x + (size_t)blockIdx.x * (IMGS * 784),
                    (unsigned int)(uintptr_t)&u.a.m.xr[0], IMGS * 784);
    }
    {
        float4 z4 = make_float4(0.f, 0.f, 0.f, 0.f);
        float4* xz = (float4*)&u.a.xs[0][0][0];
        for (int i = tid; i < IMGS * 29 * 32 / 4; i += NTHREADS) xz[i] = z4;
        float4* pz = (float4*)&z1p[0][0][0][0];
        for (int i = tid; i < (int)(sizeof(z1p) / 16); i += NTHREADS) pz[i] = z4;
        for (int i = tid; i < 72;   i += NTHREADS) w1s[i]  = W1[i];
        for (int i = tid; i < 1152; i += NTHREADS) w2s[i]  = W2[i];
        for (int i = tid; i < 784;  i += NTHREADS) wfcs[i] = Wfc[i];
        if (tid < 8)    b1s[tid] = b1[tid];
        if (tid < 16)   b2s[tid] = b2[tid];
        if (tid == 0)   bfcs = bfc[0];
        if (tid < IMGS) partial[tid] = 0.0f;
    }
    if (wave == 0) __builtin_amdgcn_s_wait_tensorcnt(0);   // TDM done (wave0's counter)
    __syncthreads();

    // ---- Phase 0b: copy TDM-landed rows into padded interior (one row/thread) ----
    for (int i = tid; i < IMGS * 28; i += NTHREADS) {
        int img = i / 28, iy = i % 28;
        const float4* s = (const float4*)&u.a.m.xr[img * 784 + iy * 28];
        float* d = &u.a.xs[img][iy + 1][1];
        #pragma unroll
        for (int q = 0; q < 7; ++q) {
            float4 v = s[q];
            d[q * 4 + 0] = v.x; d[q * 4 + 1] = v.y;
            d[q * 4 + 2] = v.z; d[q * 4 + 3] = v.w;
        }
    }
    __syncthreads();

    // ---- Phase 1: conv1 im2col, one row per thread (vectorized stores) ----
    for (int i = tid; i < IMGS * 196; i += NTHREADS) {
        int img = i / 196, p = i % 196;
        int oy = p / 14, ox = p % 14;
        const float* src = &u.a.xs[img][2 * oy][2 * ox];   // pad absorbs the -1
        float v[9];
        #pragma unroll
        for (int ky = 0; ky < 3; ++ky)
            #pragma unroll
            for (int kx = 0; kx < 3; ++kx)
                v[ky * 3 + kx] = src[ky * 32 + kx];
        v8bf lo, hi;
        #pragma unroll
        for (int j = 0; j < 8; ++j) lo[j] = (__bf16)v[j];
        hi[0] = (__bf16)v[8];
        #pragma unroll
        for (int j = 1; j < 8; ++j) hi[j] = (__bf16)0.0f;
        __bf16* drow = &u.a.m.im1[(img * 208 + p) * 16];
        *(v8bf*)drow       = lo;
        *(v8bf*)(drow + 8) = hi;
    }
    __syncthreads();

    // ---- Phase 2: conv1 WMMA (52 tiles), packed channel-major z1 stores ----
    {
        V16 B1;   // B[k][n] = W1[n][0][k], k<9, n<8
        #pragma unroll
        for (int j = 0; j < 8; ++j) {
            int k = hiK + j;
            B1.h[0][j] = (__bf16)((n < 8 && k < 9) ? w1s[n * 9 + k] : 0.0f);
            B1.h[1][j] = (__bf16)0.0f;                 // K 16..31
        }
        v8bf zero8;
        #pragma unroll
        for (int j = 0; j < 8; ++j) zero8[j] = (__bf16)0.0f;

        for (int t = wave; t < IMGS * 13; t += NWAVES) {
            int img = t / 13, tl = t % 13;             // wave-uniform
            int row = t * 16 + (lane & 15);            // == img*208 + tl*16 + m
            V16 A;
            A.h[0] = *(const v8bf*)&u.a.m.im1[row * 16 + hiK];
            A.h[1] = zero8;                            // keep pads finite
            v8f acc = {};
            acc = __builtin_amdgcn_wmma_f32_16x16x32_bf16(
                    false, A.v, false, B1.v, (short)0, acc, false, false);
            if (n < 8) {                               // lane column = channel
                v8bf o;
                #pragma unroll
                for (int r = 0; r < 8; ++r)
                    o[r] = (__bf16)fmaxf(acc[r] + b1s[n], 0.0f);
                *(v8bf*)&u.a.z1[img][n][tl * 16 + hiM] = o;   // 8 contiguous pixels
            }
        }
    }
    __syncthreads();

    // ---- Phase 3: transpose z1 (channel-major) -> z1p (channel-last, padded) ----
    for (int i = tid; i < IMGS * 196; i += NTHREADS) {
        int img = i / 196, p = i % 196;
        int oy = p / 14, ox = p % 14;
        v8bf pk;
        #pragma unroll
        for (int c = 0; c < 8; ++c) pk[c] = u.a.z1[img][c][p];
        *(v8bf*)&z1p[img][oy + 1][ox + 1][0] = pk;
    }
    __syncthreads();

    // ---- Phase 4: conv2 im2col, one row per thread, 9x b128 copies ----
    for (int i = tid; i < IMGS * 49; i += NTHREADS) {
        int img = i / 49, p = i % 49;
        int oy = p / 7, ox = p % 7;
        __bf16* drow = &u.im2[(img * 64 + p) * 96];
        const __bf16* sb = &z1p[img][2 * oy][2 * ox][0];   // pad absorbs the -1
        #pragma unroll
        for (int ky = 0; ky < 3; ++ky)
            #pragma unroll
            for (int kx = 0; kx < 3; ++kx)
                *(v8bf*)(drow + (ky * 3 + kx) * 8) =
                    *(const v8bf*)(sb + (ky * 16 + kx) * 8);
        v8bf z;
        #pragma unroll
        for (int j = 0; j < 8; ++j) z[j] = (__bf16)0.0f;
        *(v8bf*)(drow + 72) = z;                           // zero K pads 72..95
        *(v8bf*)(drow + 80) = z;
        *(v8bf*)(drow + 88) = z;
    }
    __syncthreads();

    // ---- Phase 5: conv2 WMMA (16 tiles x 3 K-steps) + fused FC reduction ----
    {
        V16 B2[3];   // B[k][n], k = (ky*3+kx)*8 + c  ->  W2[n][c][ky][kx]
        #pragma unroll
        for (int ks = 0; ks < 3; ++ks) {
            #pragma unroll
            for (int jj = 0; jj < 8; ++jj) {
                int k0 = ks * 32 + hiK + jj;
                int k1 = ks * 32 + 16 + hiK + jj;
                float v0 = 0.f, v1 = 0.f;
                if (k0 < 72) { int j = k0 >> 3, c = k0 & 7; v0 = w2s[n * 72 + c * 9 + j]; }
                if (k1 < 72) { int j = k1 >> 3, c = k1 & 7; v1 = w2s[n * 72 + c * 9 + j]; }
                B2[ks].h[0][jj] = (__bf16)v0;
                B2[ks].h[1][jj] = (__bf16)v1;
            }
        }

        for (int t = wave; t < IMGS * 4; t += NWAVES) {
            int img = t >> 2, tl = t & 3;              // wave-uniform, no div
            int row = t * 16 + (lane & 15);            // == img*64 + tl*16 + m
            const __bf16* rp = &u.im2[row * 96];
            v8f acc = {};
            #pragma unroll
            for (int ks = 0; ks < 3; ++ks) {
                V16 A;
                A.h[0] = *(const v8bf*)(rp + ks * 32 + hiK);
                A.h[1] = *(const v8bf*)(rp + ks * 32 + 16 + hiK);
                acc = __builtin_amdgcn_wmma_f32_16x16x32_bf16(
                        false, A.v, false, B2[ks].v, (short)0, acc, false, false);
            }
            float fsum = 0.0f;
            #pragma unroll
            for (int r = 0; r < 8; ++r) {
                int p = tl * 16 + r + hiM;
                if (p < 49) {
                    float v = fmaxf(acc[r] + b2s[n], 0.0f);
                    fsum += v * wfcs[n * 49 + p];      // flat index = n*49 + p
                }
            }
            atomicAdd(&partial[img], fsum);            // img uniform: 1 atomic/lane/tile
        }
    }
    __syncthreads();

    // ---- Phase 6: softplus head ----
    if (tid < IMGS) {
        float s = partial[tid] + bfcs;
        float sp = fmaxf(s, 0.0f) + log1pf(expf(-fabsf(s)));   // stable softplus
        float rate = sp + 0.001f;
        float prob = 1.0f - expf(-rate);
        prob = fminf(fmaxf(prob, 1e-6f), 1.0f - 1e-6f);
        out[(size_t)blockIdx.x * IMGS + tid] = prob;
    }
}

extern "C" void kernel_launch(void* const* d_in, const int* in_sizes, int n_in,
                              void* d_out, int out_size, void* d_ws, size_t ws_size,
                              hipStream_t stream) {
    const float* x   = (const float*)d_in[0];
    const float* W1  = (const float*)d_in[1];
    const float* b1  = (const float*)d_in[2];
    const float* W2  = (const float*)d_in[3];
    const float* b2  = (const float*)d_in[4];
    const float* Wfc = (const float*)d_in[5];
    const float* bfc = (const float*)d_in[6];
    float* out = (float*)d_out;

    const int B = in_sizes[0] / 784;          // 65536
    const int grid = B / IMGS;                // divisible for this workload
    diffsol_fused_kernel<<<grid, NTHREADS, 0, stream>>>(x, W1, b1, W2, b2, Wfc, bfc, out);
}
